// DualParallelAttention_88218628260361
// MI455X (gfx1250) — matrix operand order repaired and checked by reading, not compiled, not run on hardware
//
#include <hip/hip_runtime.h>

typedef __attribute__((ext_vector_type(16))) _Float16 v16h;
typedef __attribute__((ext_vector_type(8)))  float    v8f;

#define DN  256
#define DE  128
#define NH  8
#define DHN 32
#define DHE 16

// ---------------------------------------------------------------- utilities

__global__ __launch_bounds__(256) void fill_kernel(float* p, float v, long n) {
  long i = (long)blockIdx.x * blockDim.x + threadIdx.x;
  if (i < n) p[i] = v;
}

__device__ inline void atomicMaxF(float* p, float v) {
  // sign-aware monotonic trick: valid for any init (we init to -1e9)
  if (v >= 0.0f) atomicMax((int*)p, __float_as_int(v));
  else           atomicMin((unsigned int*)p, __float_as_uint(v));
}

__device__ inline float block_sum(float v, float* sbuf, int D, int t) {
  sbuf[t] = v;
  __syncthreads();
  for (int off = D >> 1; off > 0; off >>= 1) {
    if (t < off) sbuf[t] += sbuf[t + off];
    __syncthreads();
  }
  float r = sbuf[0];
  __syncthreads();
  return r;
}

// ------------------------------------------------- WMMA GEMM (f16x3 split)
// C[M,Nout] (+)= A[M,K] @ W[K,Nout] + bias.  One wave per 16x32 output tile.
// fp32 operands split into f16 hi/lo; D = Ahi*Bhi + Alo*Bhi + Ahi*Blo with
// f32 accumulation -> ~2^-22 relative error (fp32-comparable).
// M-tail: A row pointer clamped to M-1 (loads unconditional; WMMA rows are
// independent and out-of-range D rows are simply not stored).
__global__ __launch_bounds__(32) void gemm_wmma_kernel(
    const float* __restrict__ A, const float* __restrict__ W,
    const float* __restrict__ bias, float* __restrict__ C,
    int M, int K, int Nout, int accum) {
  int lane = threadIdx.x;
  int m0 = blockIdx.x << 4;
  int n0 = blockIdx.y << 5;                       // 32 output columns per wave
  int mrow = m0 + (lane & 15);
  int arow = (mrow < M) ? mrow : (M - 1);         // clamp: OOB rows unused
  int ncol = n0 + (lane & 15);
  int kA = (lane >> 4) << 3;                      // A frag K-base per lane half
  int kB = (lane >> 4) << 3;                      // B frag K-base per lane half
  const float4* Arow = (const float4*)(A + (size_t)arow * K + kA);
  const float*  Wc0  = W + ncol;
  const float*  Wc1  = W + ncol + 16;
  v8f acc0 = {};
  v8f acc1 = {};
  for (int k0 = 0; k0 < K; k0 += 32) {
    const float4* ap = Arow + (k0 >> 2);
    if (k0 + 32 < K) __builtin_prefetch(ap + 8, 0, 3);
    // A frag: K = kA + {0..7} (elements 0-7) and kA + {16..23} (elements 8-15)
    float4 a0 = ap[0], a1 = ap[1], a2 = ap[4], a3 = ap[5];
    float av[16] = {a0.x, a0.y, a0.z, a0.w, a1.x, a1.y, a1.z, a1.w,
                    a2.x, a2.y, a2.z, a2.w, a3.x, a3.y, a3.z, a3.w};
    v16h ah, al;
#pragma unroll
    for (int e2 = 0; e2 < 16; ++e2) {
      _Float16 h = (_Float16)av[e2];
      ah[e2] = h;
      al[e2] = (_Float16)(av[e2] - (float)h);
    }
    v16h bh0, bl0, bh1, bl1;
#pragma unroll
    for (int e2 = 0; e2 < 16; ++e2) {
      size_t ro = (size_t)(k0 + kB + e2 + ((e2 >> 3) << 3)) * Nout;
      float b0 = Wc0[ro];
      float b1 = Wc1[ro];
      _Float16 h0 = (_Float16)b0;
      bh0[e2] = h0;
      bl0[e2] = (_Float16)(b0 - (float)h0);
      _Float16 h1 = (_Float16)b1;
      bh1[e2] = h1;
      bl1[e2] = (_Float16)(b1 - (float)h1);
    }
    acc0 = __builtin_amdgcn_wmma_f32_16x16x32_f16(false, ah, false, bh0, (short)0, acc0, false, false);
    acc0 = __builtin_amdgcn_wmma_f32_16x16x32_f16(false, al, false, bh0, (short)0, acc0, false, false);
    acc0 = __builtin_amdgcn_wmma_f32_16x16x32_f16(false, ah, false, bl0, (short)0, acc0, false, false);
    acc1 = __builtin_amdgcn_wmma_f32_16x16x32_f16(false, ah, false, bh1, (short)0, acc1, false, false);
    acc1 = __builtin_amdgcn_wmma_f32_16x16x32_f16(false, al, false, bh1, (short)0, acc1, false, false);
    acc1 = __builtin_amdgcn_wmma_f32_16x16x32_f16(false, ah, false, bl1, (short)0, acc1, false, false);
  }
  float bv0 = bias ? bias[ncol] : 0.0f;
  float bv1 = bias ? bias[ncol + 16] : 0.0f;
#pragma unroll
  for (int r = 0; r < 8; ++r) {
    int mm = m0 + r + ((lane >> 4) << 3);         // C frag: M = r + 8*lanehalf
    if (mm < M) {
      size_t o = (size_t)mm * Nout + ncol;
      float v0 = acc0[r] + bv0;
      float v1 = acc1[r] + bv1;
      if (accum) { v0 += C[o]; v1 += C[o + 16]; }
      C[o] = v0;
      C[o + 16] = v1;
    }
  }
}

// ------------------------------------------------- small-N (H=8) projection
__global__ __launch_bounds__(256) void gemm_n8_kernel(
    const float* __restrict__ A, const float* __restrict__ W,
    const float* __restrict__ bias, float* __restrict__ C, long M, int K) {
  long i = (long)blockIdx.x * blockDim.x + threadIdx.x;
  if (i >= M * NH) return;
  long row = i >> 3; int h = (int)(i & 7);
  float s = bias ? bias[h] : 0.0f;
  const float4* a = (const float4*)(A + row * (size_t)K);
  for (int k = 0; k < (K >> 2); ++k) {
    float4 av = a[k];
    s = fmaf(av.x, W[(size_t)(4 * k + 0) * NH + h], s);
    s = fmaf(av.y, W[(size_t)(4 * k + 1) * NH + h], s);
    s = fmaf(av.z, W[(size_t)(4 * k + 2) * NH + h], s);
    s = fmaf(av.w, W[(size_t)(4 * k + 3) * NH + h], s);
  }
  C[i] = s;
}

// ------------------------------------------------- attention pieces
__global__ __launch_bounds__(256) void attn_score_kernel(
    const float* __restrict__ Q, const float* __restrict__ Km,
    const float* __restrict__ biasv, const float* __restrict__ logt,
    const int* __restrict__ sidx, const int* __restrict__ tidx,
    float* __restrict__ sout, float* __restrict__ smax,
    long NE2, int D, int dh, float scale, int bias_by_src) {
  long i = (long)blockIdx.x * blockDim.x + threadIdx.x;
  if (i >= NE2 * NH) return;
  long e = i >> 3; int h = (int)(i & 7);
  int s_ = sidx[e], t_ = tidx[e];
  const float4* q = (const float4*)(Q  + (size_t)t_ * D + h * dh);
  const float4* k = (const float4*)(Km + (size_t)s_ * D + h * dh);
  float d = 0.0f;
  for (int j = 0; j < (dh >> 2); ++j) {
    float4 qv = q[j], kv = k[j];
    d = fmaf(qv.x, kv.x, d);
    d = fmaf(qv.y, kv.y, d);
    d = fmaf(qv.z, kv.z, d);
    d = fmaf(qv.w, kv.w, d);
  }
  float bvv = bias_by_src ? biasv[(size_t)s_ * NH + h] : biasv[i];
  float sc = (d * scale + bvv) * expf(logt[h]);
  sout[i] = sc;
  atomicMaxF(&smax[(size_t)t_ * NH + h], sc);
}

__global__ __launch_bounds__(256) void attn_exp_kernel(
    const int* __restrict__ tidx, float* __restrict__ sbuf,
    const float* __restrict__ smax, float* __restrict__ sden, long NE2) {
  long i = (long)blockIdx.x * blockDim.x + threadIdx.x;
  if (i >= NE2 * NH) return;
  long e = i >> 3; int h = (int)(i & 7);
  int t_ = tidx[e];
  float ex = expf(sbuf[i] - smax[(size_t)t_ * NH + h]);
  sbuf[i] = ex;
  atomicAdd(&sden[(size_t)t_ * NH + h], ex);
}

__global__ void attn_agg_kernel(
    const float* __restrict__ V, const int* __restrict__ sidx,
    const int* __restrict__ tidx, const float* __restrict__ sbuf,
    const float* __restrict__ sden, float* __restrict__ agg, int D, int dh) {
  long e = blockIdx.x; int t = threadIdx.x;
  int s_ = sidx[e], tg = tidx[e];
  int h = t / dh;
  float den = sden[(size_t)tg * NH + h] + 1e-10f;
  float w = sbuf[e * NH + h] / den;
  atomicAdd(&agg[(size_t)tg * D + t], V[(size_t)s_ * D + t] * w);
}

__global__ __launch_bounds__(256) void edge_ctx_kernel(
    const float* __restrict__ nc1, const float* __restrict__ nc2,
    const float* __restrict__ bctx, const int* __restrict__ src,
    const int* __restrict__ tgt, float* __restrict__ cbE, long E) {
  long i = (long)blockIdx.x * blockDim.x + threadIdx.x;
  if (i >= E * NH) return;
  long e = i >> 3; int h = (int)(i & 7);
  cbE[i] = nc1[(size_t)src[e] * NH + h] + nc2[(size_t)tgt[e] * NH + h] + bctx[h];
}

// ------------------------------------------------- LayerNorm (LDS reduce)
__global__ void resid_ln_kernel(
    const float* __restrict__ x, const float* __restrict__ pre,
    const float* __restrict__ g, const float* __restrict__ b,
    float* __restrict__ out, int D) {
  __shared__ float sbuf[256];
  long row = blockIdx.x; int t = threadIdx.x;
  size_t o = row * (size_t)D + t;
  float y = x[o] + pre[o];
  float m = block_sum(y, sbuf, D, t) / (float)D;
  float d = y - m;
  float v = block_sum(d * d, sbuf, D, t) / (float)D;
  out[o] = d * rsqrtf(v + 1e-5f) * g[t] + b[t];
}

// bridge: new_edges = LN(edge_out + tmpE + p1[src] + p2[tgt]) + scatter sums
__global__ __launch_bounds__(128) void bridge_edge_kernel(
    const float* __restrict__ edge_out, const float* __restrict__ tmpE,
    const float* __restrict__ p1, const float* __restrict__ p2,
    const int* __restrict__ src, const int* __restrict__ tgt,
    const float* __restrict__ g, const float* __restrict__ b,
    float* __restrict__ out_edges, float* __restrict__ esum,
    float* __restrict__ ecnt) {
  __shared__ float sbuf[128];
  long e = blockIdx.x; int t = threadIdx.x;
  int s_ = src[e], tg = tgt[e];
  size_t o = e * (size_t)DE + t;
  float y = edge_out[o] + tmpE[o] + p1[(size_t)s_ * DE + t] + p2[(size_t)tg * DE + t];
  float m = block_sum(y, sbuf, DE, t) / (float)DE;
  float d = y - m;
  float v = block_sum(d * d, sbuf, DE, t) / (float)DE;
  float ne = d * rsqrtf(v + 1e-5f) * g[t] + b[t];
  out_edges[o] = ne;
  atomicAdd(&esum[(size_t)s_ * DE + t], ne);
  atomicAdd(&esum[(size_t)tg * DE + t], ne);
  if (t == 0) {
    atomicAdd(&ecnt[s_], 1.0f);
    atomicAdd(&ecnt[tg], 1.0f);
  }
}

__global__ __launch_bounds__(256) void emean_kernel(
    float* __restrict__ esum, const float* __restrict__ ecnt, long N) {
  long i = (long)blockIdx.x * blockDim.x + threadIdx.x;
  if (i >= N * DE) return;
  esum[i] = esum[i] / (ecnt[i >> 7] + 1e-10f);
}

// ---------------------------------------------------------------- host side

static inline void launch_gemm(const float* A, const float* W, const float* bias,
                               float* C, long M, int K, int Nout, int accum,
                               hipStream_t s) {
  dim3 grid((unsigned)((M + 15) / 16), (unsigned)(Nout / 32));
  gemm_wmma_kernel<<<grid, 32, 0, s>>>(A, W, bias, C, (int)M, K, Nout, accum);
}

extern "C" void kernel_launch(void* const* d_in, const int* in_sizes, int n_in,
                              void* d_out, int out_size, void* d_ws, size_t ws_size,
                              hipStream_t stream) {
  (void)n_in; (void)out_size; (void)ws_size;
  const float* x    = (const float*)d_in[0];
  const float* ef   = (const float*)d_in[1];
  const int*   ei   = (const int*)d_in[2];
  const int*   ea   = (const int*)d_in[3];
  const float* n_Wq = (const float*)d_in[4],  *n_bq  = (const float*)d_in[5];
  const float* n_Wk = (const float*)d_in[6],  *n_bk  = (const float*)d_in[7];
  const float* n_Wv = (const float*)d_in[8],  *n_bv  = (const float*)d_in[9];
  const float* n_Web= (const float*)d_in[10], *n_beb = (const float*)d_in[11];
  const float* n_Wo = (const float*)d_in[12], *n_bo  = (const float*)d_in[13];
  const float* n_logt = (const float*)d_in[14];
  const float* n_g  = (const float*)d_in[15], *n_b   = (const float*)d_in[16];
  const float* e_Wq = (const float*)d_in[17], *e_bq  = (const float*)d_in[18];
  const float* e_Wk = (const float*)d_in[19], *e_bk  = (const float*)d_in[20];
  const float* e_Wv = (const float*)d_in[21], *e_bv  = (const float*)d_in[22];
  const float* e_Wctx=(const float*)d_in[23], *e_bctx= (const float*)d_in[24];
  const float* e_Wo = (const float*)d_in[25], *e_bo  = (const float*)d_in[26];
  const float* e_logt = (const float*)d_in[27];
  const float* e_g  = (const float*)d_in[28], *e_b   = (const float*)d_in[29];
  const float* r_We = (const float*)d_in[30], *r_be  = (const float*)d_in[31];
  const float* r_eg = (const float*)d_in[32], *r_ebt = (const float*)d_in[33];
  const float* r_Wn = (const float*)d_in[34], *r_bn  = (const float*)d_in[35];
  const float* r_ng = (const float*)d_in[36], *r_nbt = (const float*)d_in[37];

  const long N  = in_sizes[0] / DN;
  const long E  = in_sizes[1] / DE;
  const long EA = in_sizes[3] / 2;
  const int* srcN = ei;      const int* tgtN = ei + E;
  const int* seA  = ea;      const int* teA  = ea + EA;

  float* out_nodes = (float*)d_out;
  float* out_edges = out_nodes + (size_t)N * DN;

  float* WS = (float*)d_ws;
  const size_t nDN = (size_t)N * DN;
  const size_t eDE = (size_t)E * DE;
  const size_t nH  = (size_t)N * NH;
  const size_t eH  = (size_t)E * NH;

  float* node_out = WS;                // live A -> C
  float* edge_out = WS + nDN;          // live B -> C
  float* pool     = WS + nDN + eDE;    // stage-reused scratch

  auto fill = [&](float* p, float v, size_t n) {
    fill_kernel<<<(unsigned)((n + 255) / 256), 256, 0, stream>>>(p, v, (long)n);
  };
  const unsigned gEH  = (unsigned)((E * NH + 255) / 256);
  const unsigned gAH  = (unsigned)((EA * NH + 255) / 256);
  const unsigned gNH8 = (unsigned)((N * NH + 255) / 256);

  // ================= Stage A: NodeAttention =================
  {
    float* Qn    = pool;
    float* Kn    = Qn + nDN;
    float* Vn    = Kn + nDN;
    float* aggN  = Vn + nDN;
    float* eb    = aggN + nDN;
    float* w1    = eb + eH;
    float* smax1 = w1 + eH;
    float* sden1 = smax1 + nH;
    float* tmpA  = Qn;  // reuse after scores/agg done

    fill(aggN, 0.0f, nDN);
    fill(smax1, -1e9f, nH);
    fill(sden1, 0.0f, nH);
    launch_gemm(x, n_Wq, n_bq, Qn, N, DN, DN, 0, stream);
    launch_gemm(x, n_Wk, n_bk, Kn, N, DN, DN, 0, stream);
    launch_gemm(x, n_Wv, n_bv, Vn, N, DN, DN, 0, stream);
    gemm_n8_kernel<<<gEH, 256, 0, stream>>>(ef, n_Web, n_beb, eb, E, DE);
    attn_score_kernel<<<gEH, 256, 0, stream>>>(Qn, Kn, eb, n_logt, srcN, tgtN,
        w1, smax1, E, DN, DHN, 0.17677669529663689f /*1/sqrt(32)*/, 0);
    attn_exp_kernel<<<gEH, 256, 0, stream>>>(tgtN, w1, smax1, sden1, E);
    attn_agg_kernel<<<(unsigned)E, DN, 0, stream>>>(Vn, srcN, tgtN, w1, sden1,
                                                    aggN, DN, DHN);
    launch_gemm(aggN, n_Wo, n_bo, tmpA, N, DN, DN, 0, stream);
    resid_ln_kernel<<<(unsigned)N, DN, 0, stream>>>(x, tmpA, n_g, n_b,
                                                    node_out, DN);
  }

  // ================= Stage B: EdgeAttention =================
  {
    float* Qe    = pool;
    float* Ke    = Qe + eDE;
    float* Ve    = Ke + eDE;
    float* agg2  = Ve + eDE;
    float* nc1   = agg2 + eDE;
    float* nc2   = nc1 + nH;
    float* cbE   = nc2 + nH;
    float* smax2 = cbE + eH;
    float* sden2 = smax2 + eH;
    float* w2    = sden2 + eH;   // EA*NH
    float* tmpB  = Qe;           // reuse after scores done

    fill(agg2, 0.0f, eDE);
    fill(smax2, -1e9f, eH);
    fill(sden2, 0.0f, eH);
    launch_gemm(ef, e_Wq, e_bq, Qe, E, DE, DE, 0, stream);
    launch_gemm(ef, e_Wk, e_bk, Ke, E, DE, DE, 0, stream);
    launch_gemm(ef, e_Wv, e_bv, Ve, E, DE, DE, 0, stream);
    // ctx@e_Wctx decomposed: per-node projections + per-edge gather-add
    gemm_n8_kernel<<<gNH8, 256, 0, stream>>>(x, e_Wctx, nullptr, nc1, N, DN);
    gemm_n8_kernel<<<gNH8, 256, 0, stream>>>(x, e_Wctx + (size_t)DN * NH,
                                             nullptr, nc2, N, DN);
    edge_ctx_kernel<<<gEH, 256, 0, stream>>>(nc1, nc2, e_bctx, srcN, tgtN, cbE, E);
    attn_score_kernel<<<gAH, 256, 0, stream>>>(Qe, Ke, cbE, e_logt, seA, teA,
        w2, smax2, EA, DE, DHE, 0.25f /*1/sqrt(16)*/, 1);
    attn_exp_kernel<<<gAH, 256, 0, stream>>>(teA, w2, smax2, sden2, EA);
    attn_agg_kernel<<<(unsigned)EA, DE, 0, stream>>>(Ve, seA, teA, w2, sden2,
                                                     agg2, DE, DHE);
    launch_gemm(agg2, e_Wo, e_bo, tmpB, E, DE, DE, 0, stream);
    resid_ln_kernel<<<(unsigned)E, DE, 0, stream>>>(ef, tmpB, e_g, e_b,
                                                    edge_out, DE);
  }

  // ================= Stage C: ReconciliationBridge =================
  {
    const size_t nDE = (size_t)N * DE;
    float* p1   = pool;
    float* p2   = p1 + nDE;
    float* tmpE = p2 + nDE;
    float* esum = tmpE + eDE;
    float* ecnt = esum + nDE;
    float* tmpN = ecnt + N;

    fill(esum, 0.0f, nDE);
    fill(ecnt, 0.0f, (size_t)N);
    // r_We rows: [0:128)=edge part, [128:384)=src-node part, [384:640)=tgt part
    launch_gemm(node_out, r_We + (size_t)DE * DE,        nullptr, p1, N, DN, DE, 0, stream);
    launch_gemm(node_out, r_We + (size_t)(DE + DN) * DE, nullptr, p2, N, DN, DE, 0, stream);
    launch_gemm(edge_out, r_We, r_be, tmpE, E, DE, DE, 0, stream);
    bridge_edge_kernel<<<(unsigned)E, DE, 0, stream>>>(edge_out, tmpE, p1, p2,
        srcN, tgtN, r_eg, r_ebt, out_edges, esum, ecnt);
    emean_kernel<<<(unsigned)((nDE + 255) / 256), 256, 0, stream>>>(esum, ecnt, N);
    // r_Wn rows: [0:256)=node_out part, [256:384)=emean part (accumulate)
    launch_gemm(node_out, r_Wn, r_bn, tmpN, N, DN, DN, 0, stream);
    launch_gemm(esum, r_Wn + (size_t)DN * DN, nullptr, tmpN, N, DE, DN, 1, stream);
    resid_ln_kernel<<<(unsigned)N, DN, 0, stream>>>(node_out, tmpN, r_ng, r_nbt,
                                                    out_nodes, DN);
  }
}
